// EnterpriseGNN_8160437862929
// MI455X (gfx1250) — matrix operand order repaired and checked by reading, not compiled, not run on hardware
//
#include <hip/hip_runtime.h>
#include <hip/hip_bf16.h>

typedef __attribute__((ext_vector_type(16))) _Float16 v16h;
typedef __attribute__((ext_vector_type(8)))  float    v8f;

#define F_IN  128
#define F1    32
#define F2    16

// ---------------- degree / symmetric norm ----------------
__global__ void k_deg_init(float* __restrict__ deg, int N) {
  int i = blockIdx.x * blockDim.x + threadIdx.x;
  if (i < N) deg[i] = 1.0f;  // self-loop weight
}

__global__ void k_deg_accum(const int* __restrict__ dst, const float* __restrict__ w,
                            float* __restrict__ deg, int E) {
  int e = blockIdx.x * blockDim.x + threadIdx.x;
  if (e < E) atomicAdd(&deg[dst[e]], w[e]);
}

__global__ void k_dinv(float* __restrict__ deg, int N) {
  int i = blockIdx.x * blockDim.x + threadIdx.x;
  if (i < N) {
    float d = deg[i];
    deg[i] = (d > 0.0f) ? rsqrtf(d) : 0.0f;  // in-place deg -> dinv
  }
}

// ---------------- WMMA fragment loaders (CDNA5 16-bit layouts) ----------------
// A 16x32 f16: lanes 0-15 hold M=lane, K {0..7,16..23}; lanes 16-31 hold K {8..15,24..31}.
// Row index clamped (no divergence); tail rows are duplicates, discarded at store.
__device__ __forceinline__ v16h load_a16x32(const float* __restrict__ X, int lda,
                                            int m0, int k0, int mMax) {
  int lane  = threadIdx.x & 31;
  int m     = m0 + (lane & 15);
  m = (m < mMax) ? m : (mMax - 1);
  int khalf = (lane >> 4) * 8;
  const float* row = X + (size_t)m * lda + k0 + khalf;  // 32B aligned
  float4 u0 = *(const float4*)(row + 0);   // K = khalf+0..3
  float4 u1 = *(const float4*)(row + 4);   // K = khalf+4..7
  float4 u2 = *(const float4*)(row + 16);  // K = 16+khalf+0..3
  float4 u3 = *(const float4*)(row + 20);  // K = 16+khalf+4..7
  v16h a;
  a[0]  = (_Float16)u0.x; a[1]  = (_Float16)u0.y;
  a[2]  = (_Float16)u0.z; a[3]  = (_Float16)u0.w;
  a[4]  = (_Float16)u1.x; a[5]  = (_Float16)u1.y;
  a[6]  = (_Float16)u1.z; a[7]  = (_Float16)u1.w;
  a[8]  = (_Float16)u2.x; a[9]  = (_Float16)u2.y;
  a[10] = (_Float16)u2.z; a[11] = (_Float16)u2.w;
  a[12] = (_Float16)u3.x; a[13] = (_Float16)u3.y;
  a[14] = (_Float16)u3.z; a[15] = (_Float16)u3.w;
  return a;
}

// B 32x16 f16: lanes 0-15 => N=lane, K {0..7,16..23}; lanes 16-31 => N=lane-16, K {8..15,24..31}
__device__ __forceinline__ v16h load_b32x16(const float* __restrict__ B, int ldb,
                                            int k0, int n0) {
  int lane = threadIdx.x & 31;
  int n    = n0 + (lane & 15);
  int kb   = k0 + (lane >> 4) * 8;
  v16h b;
#pragma unroll
  for (int j = 0; j < 16; ++j)
    b[j] = (_Float16)B[(size_t)(kb + j + (j >= 8 ? 8 : 0)) * ldb + n];
  return b;
}

// C/D 16x16 f32: VGPR r, lanes 0-15 => M=r, lanes 16-31 => M=r+8; N=lane&15.
// Tail check is wave-uniform; full tiles take the unguarded path.
__device__ __forceinline__ void store_c16x16(float* __restrict__ C, int ldc,
                                             int m0, int n0, v8f c, int mMax) {
  int lane = threadIdx.x & 31;
  int n    = n0 + (lane & 15);
  int mh   = m0 + (lane >> 4) * 8;
  if (m0 + 16 <= mMax) {
#pragma unroll
    for (int r = 0; r < 8; ++r)
      C[(size_t)(mh + r) * ldc + n] = c[r];
  } else {
#pragma unroll
    for (int r = 0; r < 8; ++r)
      if (mh + r < mMax) C[(size_t)(mh + r) * ldc + n] = c[r];
  }
}

// ---------------- skinny GEMM: C[N,NCOLS] = A[N,KDIM] @ B[KDIM,NCOLS] ----------------
template <int KDIM, int NCOLS>
__global__ void k_gemm_wmma(const float* __restrict__ A, const float* __restrict__ B,
                            float* __restrict__ C, int N) {
  long long gtid = (long long)blockIdx.x * blockDim.x + threadIdx.x;
  long long wave = gtid >> 5;
  const int colTiles = NCOLS / 16;
  int rowTile = (int)(wave / colTiles);
  int colTile = (int)(wave % colTiles);
  int m0 = rowTile * 16;
  if (m0 >= N) return;  // wave-uniform: EXEC all-ones at every WMMA
  int n0 = colTile * 16;
  v8f c = {};
#pragma unroll
  for (int k0 = 0; k0 < KDIM; k0 += 32) {
    v16h a = load_a16x32(A, KDIM, m0, k0, N);
    v16h b = load_b32x16(B, NCOLS, k0, n0);
    c = __builtin_amdgcn_wmma_f32_16x16x32_f16(false, a, false, b, (short)0, c,
                                               false, false);
  }
  store_c16x16(C, NCOLS, m0, n0, c, N);
}

// ---------------- message passing ----------------
template <int FEAT>
__global__ void k_selfloop(const float* __restrict__ h, const float* __restrict__ dinv,
                           float* __restrict__ acc, int N) {
  int i = blockIdx.x * blockDim.x + threadIdx.x;
  if (i < N * FEAT) {
    float di = dinv[i / FEAT];  // FEAT is power of two -> shift
    acc[i] = h[i] * di * di;    // self-loop contribution (weight 1)
  }
}

template <int FEAT>  // FEAT lanes per edge, lane = feature
__global__ void k_edge_scatter(const int* __restrict__ src, const int* __restrict__ dst,
                               const float* __restrict__ w, const float* __restrict__ dinv,
                               const float* __restrict__ h, float* __restrict__ acc,
                               int E) {
  long long gid = (long long)blockIdx.x * blockDim.x + threadIdx.x;
  long long e = gid / FEAT;
  int f = (int)(gid & (FEAT - 1));
  if (e >= E) return;
  int s = src[e];
  int d = dst[e];
  float coef = dinv[s] * w[e] * dinv[d];
  atomicAdd(&acc[(size_t)d * FEAT + f], h[(size_t)s * FEAT + f] * coef);
}

template <int FEAT>
__global__ void k_bias_relu(float* __restrict__ acc, const float* __restrict__ bias, int N) {
  int i = blockIdx.x * blockDim.x + threadIdx.x;
  if (i < N * FEAT) {
    float v = acc[i] + bias[i & (FEAT - 1)];
    acc[i] = v > 0.0f ? v : 0.0f;
  }
}

// ---------------- head: out[N,3] = a2[N,16] @ Wout[16,3] + bout ----------------
__global__ void k_out(const float* __restrict__ a2, const float* __restrict__ Wout,
                      const float* __restrict__ bout, float* __restrict__ out, int N) {
  int n = blockIdx.x * blockDim.x + threadIdx.x;
  if (n >= N) return;
  float r0 = bout[0], r1 = bout[1], r2 = bout[2];
  const float4* row = (const float4*)(a2 + (size_t)n * 16);
#pragma unroll
  for (int q = 0; q < 4; ++q) {
    float4 v = row[q];
    float vv[4] = {v.x, v.y, v.z, v.w};
#pragma unroll
    for (int j = 0; j < 4; ++j) {
      int k = q * 4 + j;
      r0 = fmaf(vv[j], Wout[k * 3 + 0], r0);
      r1 = fmaf(vv[j], Wout[k * 3 + 1], r1);
      r2 = fmaf(vv[j], Wout[k * 3 + 2], r2);
    }
  }
  out[(size_t)n * 3 + 0] = r0;
  out[(size_t)n * 3 + 1] = r1;
  out[(size_t)n * 3 + 2] = r2;
}

extern "C" void kernel_launch(void* const* d_in, const int* in_sizes, int n_in,
                              void* d_out, int out_size, void* d_ws, size_t ws_size,
                              hipStream_t stream) {
  (void)n_in; (void)out_size; (void)ws_size;
  const float* x  = (const float*)d_in[0];
  const int*   ei = (const int*)d_in[1];
  const float* ew = (const float*)d_in[2];
  const float* W1 = (const float*)d_in[3];
  const float* b1 = (const float*)d_in[4];
  const float* W2 = (const float*)d_in[5];
  const float* b2 = (const float*)d_in[6];
  const float* Wo = (const float*)d_in[7];
  const float* bo = (const float*)d_in[8];
  float* out = (float*)d_out;

  const int E = in_sizes[2];
  const int N = in_sizes[0] / F_IN;
  const int* src = ei;
  const int* dst = ei + E;

  float* ws   = (float*)d_ws;
  float* dinv = ws;                          // N
  float* h1   = dinv + N;                    // N*32
  float* acc1 = h1   + (size_t)N * F1;       // N*32 (becomes a1)
  float* h2   = acc1 + (size_t)N * F1;       // N*16
  float* acc2 = h2   + (size_t)N * F2;       // N*16 (becomes a2)

  const int B = 256;
  // ---- degree + rsqrt norm (graph-only; shared by both layers) ----
  k_deg_init <<<(N + B - 1) / B, B, 0, stream>>>(dinv, N);
  k_deg_accum<<<(E + B - 1) / B, B, 0, stream>>>(dst, ew, dinv, E);
  k_dinv     <<<(N + B - 1) / B, B, 0, stream>>>(dinv, N);

  const int rowTiles = (N + 15) / 16;

  // ---- layer 1: h1 = x @ W1 ; aggregate ; +b1 ; relu ----
  {
    long long waves  = (long long)rowTiles * (F1 / 16);
    long long blocks = (waves * 32 + B - 1) / B;
    k_gemm_wmma<F_IN, F1><<<(int)blocks, B, 0, stream>>>(x, W1, h1, N);
  }
  k_selfloop<F1><<<(int)(((long long)N * F1 + B - 1) / B), B, 0, stream>>>(h1, dinv, acc1, N);
  k_edge_scatter<F1><<<(int)(((long long)E * F1 + B - 1) / B), B, 0, stream>>>(
      src, dst, ew, dinv, h1, acc1, E);
  k_bias_relu<F1><<<(int)(((long long)N * F1 + B - 1) / B), B, 0, stream>>>(acc1, b1, N);

  // ---- layer 2: h2 = a1 @ W2 ; aggregate ; +b2 ; relu ----
  {
    long long waves  = (long long)rowTiles * (F2 / 16);
    long long blocks = (waves * 32 + B - 1) / B;
    k_gemm_wmma<F1, F2><<<(int)blocks, B, 0, stream>>>(acc1, W2, h2, N);
  }
  k_selfloop<F2><<<(int)(((long long)N * F2 + B - 1) / B), B, 0, stream>>>(h2, dinv, acc2, N);
  k_edge_scatter<F2><<<(int)(((long long)E * F2 + B - 1) / B), B, 0, stream>>>(
      src, dst, ew, dinv, h2, acc2, E);
  k_bias_relu<F2><<<(int)(((long long)N * F2 + B - 1) / B), B, 0, stream>>>(acc2, b2, N);

  // ---- head ----
  k_out<<<(N + B - 1) / B, B, 0, stream>>>(acc2, Wo, bo, out, N);
}
